// ALiBiAttention_15006615733150
// MI455X (gfx1250) — matrix operand order repaired and checked by reading, not compiled, not run on hardware
//
#include <hip/hip_runtime.h>
#include <hip/hip_bf16.h>

#define USE_TDM 1   // 1: TENSOR_LOAD_TO_LDS staging; 0: GLOBAL_LOAD_ASYNC_TO_LDS staging

// ---------------- types ----------------
typedef __bf16 bf16_t;
typedef bf16_t v16bf __attribute__((ext_vector_type(16)));
typedef float  v8f   __attribute__((ext_vector_type(8)));
typedef unsigned int u32;
typedef u32 u32x4 __attribute__((ext_vector_type(4)));
typedef u32 u32x8 __attribute__((ext_vector_type(8)));
typedef unsigned short u16;

struct F2 { u32x4 a, b; };
__device__ __forceinline__ v16bf mk_frag(u32x4 a, u32x4 b) {
    F2 f{a, b};
    return __builtin_bit_cast(v16bf, f);
}

__device__ __forceinline__ u16 f2bf(float f) {
    u32 u = __float_as_uint(f);
    u32 r = u + 0x7fffu + ((u >> 16) & 1u);   // round-to-nearest-even
    return (u16)(r >> 16);
}

// async global -> LDS copy (CDNA5 ASYNCcnt path), 16 bytes per lane
__device__ __forceinline__ void async_g2l_b128(u32 lds_off, const void* gaddr) {
    asm volatile("global_load_async_to_lds_b128 %0, %1, off"
                 :: "v"(lds_off), "v"(gaddr) : "memory");
}

// TDM: DMA one 32x128 bf16 tile (row-major, row stride `stride_elems`) into LDS,
// hardware-padding each 32-elem (64B = 16 DWORD) row with 4 DWORDs -> LDS stride 40.
// D# groups per CDNA5 ISA; 2D tensor so groups 2/3 are NULL.
__device__ __forceinline__ void tdm_load_tile_32x128(u32 lds_off, const u16* gaddr,
                                                     u32 stride_elems) {
    unsigned long long ga = (unsigned long long)(uintptr_t)gaddr;
    u32x4 g0;
    g0[0] = 1u;                                             // count=1 (valid user D#)
    g0[1] = lds_off;                                        // LDS byte address
    g0[2] = (u32)ga;                                        // global_addr[31:0]
    g0[3] = (u32)((ga >> 32) & 0x01ffffffu) | (2u << 30);   // global_addr[56:32] | type=2
    u32x8 g1;
    g1[0] = (1u << 16)        // data_size = 2 bytes
          | (1u << 20)        // pad_enable
          | (3u << 22)        // pad_interval: 16 DWORDs (one 64B row)
          | (3u << 25);       // pad_amount: 4 DWORDs (16B)
    g1[1] = (32u << 16);                       // tensor_dim0 = 32 (lo16); atomic_addr=0
    g1[2] = (128u << 16);                      // tensor_dim0 hi=0 | tensor_dim1 = 128 (lo16)
    g1[3] = (32u << 16);                       // tensor_dim1 hi=0 | tile_dim0 = 32
    g1[4] = 128u;                              // tile_dim1 = 128, tile_dim2 = 0
    g1[5] = stride_elems;                      // tensor_dim0_stride[31:0] (elems)
    g1[6] = 0u;                                // stride hi | dim1_stride lo (unused, 2D)
    g1[7] = 0u;
    asm volatile("tensor_load_to_lds %0, %1" :: "s"(g0), "s"(g1) : "memory");
}

#define NQ   2048
#define CDIM 1024
#define NH   16
#define HD   64
#define BATCH 2
#define WINDOW 256
#define SCALE 0.125f

// ---------------- f32 -> bf16 conversion (8 elems/thread) ----------------
__global__ void cvt_f32_bf16(const float* __restrict__ in, u16* __restrict__ out, int n8) {
    int i = blockIdx.x * blockDim.x + threadIdx.x;
    if (i >= n8) return;
    const float4* p = (const float4*)(in + (size_t)i * 8);
    float4 a = p[0], b = p[1];
    u32x4 o;
    o[0] = (u32)f2bf(a.x) | ((u32)f2bf(a.y) << 16);
    o[1] = (u32)f2bf(a.z) | ((u32)f2bf(a.w) << 16);
    o[2] = (u32)f2bf(b.x) | ((u32)f2bf(b.y) << 16);
    o[3] = (u32)f2bf(b.z) | ((u32)f2bf(b.w) << 16);
    *(u32x4*)(out + (size_t)i * 8) = o;
}

// ---------------- WMMA GEMM: out[M,N] = A(bf16)[M,K] @ W(bf16)[N,K]^T + bias ----------------
// mode 1: N==3072, scatter to q/k/vT (bf16).  mode 2: N==1024, write f32 d_out.
// Double-buffered LDS staging: tile k+1 is DMA'd (TDM) or async-loaded while the
// 8 WMMAs of tile k execute.
#define GB_M 128
#define GB_N 128
#define GB_K 32
#define LDSTR 40   // padded LDS row stride (bf16 elems) == TDM pad output

__global__ __launch_bounds__(256)
void gemm_wmma_bf16(const u16* __restrict__ A, const u16* __restrict__ W,
                    const float* __restrict__ bias, int K, int mode,
                    float* __restrict__ out_f32,
                    u16* __restrict__ qout, u16* __restrict__ kout, u16* __restrict__ vtout) {
    __shared__ __attribute__((aligned(16))) u16 As[2][GB_M * LDSTR];
    __shared__ __attribute__((aligned(16))) u16 Ws[2][GB_N * LDSTR];

    const int tid  = threadIdx.x;
    const int lane = tid & 31;
    const int wave = tid >> 5;
    const int bm = blockIdx.y * GB_M;
    const int bn = blockIdx.x * GB_N;
    const int wm = (wave & 3) * 32;   // 32 rows per wave
    const int wn = (wave >> 2) * 64;  // 64 cols per wave

    const int mrow = lane & 15;
    const int half = lane >> 4;
    const int kbA = half * 8;    // A-frag: {kb..kb+7, kb+16..kb+23}
    const int kcB = half * 8;    // B-frag: {kb..kb+7, kb+16..kb+23}

    v8f acc[2][4];
#pragma unroll
    for (int i = 0; i < 2; i++)
#pragma unroll
        for (int j = 0; j < 4; j++)
#pragma unroll
            for (int e = 0; e < 8; e++) acc[i][j][e] = 0.0f;

#if USE_TDM
    // one wave drives the Tensor Data Mover: 2 DMA ops stage both 32x128 tiles
#define ISSUE_STAGE(nb, kk)                                                          \
    if (wave == 0) {                                                                 \
        tdm_load_tile_32x128((u32)(uintptr_t)&As[nb][0], &A[(size_t)bm * K + (kk)], (u32)K); \
        tdm_load_tile_32x128((u32)(uintptr_t)&Ws[nb][0], &W[(size_t)bn * K + (kk)], (u32)K); \
    }
#define WAIT_PREV()  if (wave == 0) __builtin_amdgcn_s_wait_tensorcnt(2)
#define WAIT_LAST()  if (wave == 0) __builtin_amdgcn_s_wait_tensorcnt(0)
#else
    // each thread owns 2 chunk ids -> issues exactly 4 async b128 per stage
#define ISSUE_STAGE(nb, kk)                                                         \
    {                                                                               \
        _Pragma("unroll")                                                           \
        for (int c = tid; c < 512; c += 256) {                                      \
            int r_ = c >> 2, q8_ = (c & 3) * 8;                                     \
            u32 la_ = (u32)(uintptr_t)&As[nb][r_ * LDSTR + q8_];                    \
            u32 lw_ = (u32)(uintptr_t)&Ws[nb][r_ * LDSTR + q8_];                    \
            async_g2l_b128(la_, &A[(size_t)(bm + r_) * K + (kk) + q8_]);            \
            async_g2l_b128(lw_, &W[(size_t)(bn + r_) * K + (kk) + q8_]);            \
        }                                                                           \
    }
#define WAIT_PREV()  asm volatile("s_wait_asynccnt 0x4" ::: "memory")
#define WAIT_LAST()  asm volatile("s_wait_asynccnt 0x0" ::: "memory")
#endif

    const int nk = K / GB_K;
    ISSUE_STAGE(0, 0);

    for (int it = 0; it < nk; ++it) {
        const int cur = it & 1;
        if (it) __syncthreads();               // everyone done reading buf[cur^1]
        if (it + 1 < nk) {
            ISSUE_STAGE(cur ^ 1, (it + 1) * GB_K);
            WAIT_PREV();                       // buf[cur] landed; next pair in flight
        } else {
            WAIT_LAST();
        }
        __syncthreads();                       // buf[cur] visible to all waves

        v16bf af[2], bf[4];
#pragma unroll
        for (int i = 0; i < 2; i++) {
            const u16* p = &As[cur][(wm + 16 * i + mrow) * LDSTR];
            af[i] = mk_frag(*(const u32x4*)(p + kbA), *(const u32x4*)(p + kbA + 16));
        }
#pragma unroll
        for (int j = 0; j < 4; j++) {
            const u16* p = &Ws[cur][(wn + 16 * j + mrow) * LDSTR];
            bf[j] = mk_frag(*(const u32x4*)(p + kcB), *(const u32x4*)(p + kcB + 16));
        }
#pragma unroll
        for (int i = 0; i < 2; i++)
#pragma unroll
            for (int j = 0; j < 4; j++)
                acc[i][j] = __builtin_amdgcn_wmma_f32_16x16x32_bf16(
                    false, af[i], false, bf[j], (short)0, acc[i][j], false, false);
    }
#undef ISSUE_STAGE
#undef WAIT_PREV
#undef WAIT_LAST

    // epilogue
    const int nn = lane & 15, hh = lane >> 4;
#pragma unroll
    for (int i = 0; i < 2; i++)
#pragma unroll
        for (int j = 0; j < 4; j++)
#pragma unroll
            for (int r = 0; r < 8; r++) {
                int gm = bm + wm + 16 * i + r + 8 * hh;
                int gn = bn + wn + 16 * j + nn;
                float v = acc[i][j][r] + bias[gn];
                if (mode == 2) {
                    out_f32[(size_t)gm * CDIM + gn] = v;
                } else {
                    int which = gn >> 10, rem = gn & 1023;
                    int h = rem >> 6, d = rem & 63;
                    int b = gm >> 11, nq = gm & 2047;
                    size_t bh = (size_t)(b * NH + h);
                    u16 bv = f2bf(v);
                    if (which == 0)      qout [(bh * NQ + nq) * HD + d] = bv;
                    else if (which == 1) kout [(bh * NQ + nq) * HD + d] = bv;
                    else                 vtout[(bh * HD + d) * NQ + nq] = bv; // V transposed
                }
            }
}

// ---------------- flash attention, sliding-window causal ALiBi ----------------
// block = 128 threads (4 waves); wave -> 16 query rows; grid = B*H * (N/64)
__global__ __launch_bounds__(128)
void attn_wmma(const u16* __restrict__ q, const u16* __restrict__ k,
               const u16* __restrict__ vt, u16* __restrict__ attn_out) {
    __shared__ __attribute__((aligned(16))) u16 P[4][16 * LDSTR];

    const int tid = threadIdx.x, lane = tid & 31, wave = tid >> 5;
    const int bh = blockIdx.x >> 5;   // 0..31 = b*16+h
    const int qc = blockIdx.x & 31;   // 64-query chunk
    const int h = bh & 15, b = bh >> 4;
    const int m0 = qc * 64 + wave * 16;
    const float slope = exp2f(-0.5f * (float)(h + 1));  // alibi slopes for H=16

    const u16* qb = q  + (size_t)bh * NQ * HD;
    const u16* kp = k  + (size_t)bh * NQ * HD;
    const u16* vb = vt + (size_t)bh * HD * NQ;

    const int nn = lane & 15, hh = lane >> 4;
    const int kbA = hh * 8, kcB = hh * 16;

    // Q fragments (K-chunks d=[0,32),[32,64)), held for whole kernel
    v16bf qf[2];
#pragma unroll
    for (int i = 0; i < 2; i++) {
        const u16* p = qb + (size_t)(m0 + nn) * HD + 32 * i;
        qf[i] = mk_frag(*(const u32x4*)(p + kbA), *(const u32x4*)(p + kbA + 16));
    }

    v8f o[4];
#pragma unroll
    for (int t = 0; t < 4; t++)
#pragma unroll
        for (int e = 0; e < 8; e++) o[t][e] = 0.0f;
    float rmax[8], rsum[8];
#pragma unroll
    for (int r = 0; r < 8; r++) { rmax[r] = -__builtin_inff(); rsum[r] = 0.0f; }

    int jlo = m0 - (WINDOW - 1); if (jlo < 0) jlo = 0;
    jlo &= ~31;
    const int jhi = m0 + 15;

    for (int j0 = jlo; j0 <= jhi; j0 += 32) {
        // ---- S = Q K^T for two 16-key tiles ----
        v8f s[2];
#pragma unroll
        for (int t = 0; t < 2; t++) {
            const u16* p = kp + (size_t)(j0 + 16 * t + nn) * HD;
            v16bf bf0 = mk_frag(*(const u32x4*)(p + kbA),      *(const u32x4*)(p + kbA + 16));
            v16bf bf1 = mk_frag(*(const u32x4*)(p + 32 + kbA), *(const u32x4*)(p + 32 + kbA + 16));
            v8f z;
#pragma unroll
            for (int e = 0; e < 8; e++) z[e] = 0.0f;
            z = __builtin_amdgcn_wmma_f32_16x16x32_bf16(false, qf[0], false, bf0, (short)0, z, false, false);
            z = __builtin_amdgcn_wmma_f32_16x16x32_bf16(false, qf[1], false, bf1, (short)0, z, false, false);
            s[t] = z;
        }

        // ---- online softmax (rows live across 16-lane halves) ----
        float alpha[8];
#pragma unroll
        for (int r = 0; r < 8; r++) {
            int iq = m0 + r + 8 * hh;
            int ja = j0 + nn, jb = j0 + 16 + nn;
            bool oka = (ja <= iq) && (iq - ja < WINDOW);
            bool okb = (jb <= iq) && (iq - jb < WINDOW);
            float va = s[0][r] * SCALE + slope * (float)(ja - iq);
            float vbv = s[1][r] * SCALE + slope * (float)(jb - iq);
            float tmax = fmaxf(oka ? va : -__builtin_inff(), okb ? vbv : -__builtin_inff());
#pragma unroll
            for (int off = 1; off <= 8; off <<= 1) tmax = fmaxf(tmax, __shfl_xor(tmax, off, 32));
            float nmax = fmaxf(rmax[r], tmax);
            float al = (nmax == -__builtin_inff()) ? 1.0f : __expf(rmax[r] - nmax);
            rmax[r] = nmax;
            float pa = oka ? __expf(va - nmax) : 0.0f;
            float pb = okb ? __expf(vbv - nmax) : 0.0f;
            float ps = pa + pb;
#pragma unroll
            for (int off = 1; off <= 8; off <<= 1) ps += __shfl_xor(ps, off, 32);
            rsum[r] = rsum[r] * al + ps;
            alpha[r] = al;
            u16* pr = &P[wave][(r + 8 * hh) * LDSTR];
            pr[nn]      = f2bf(pa);
            pr[16 + nn] = f2bf(pb);
        }
#pragma unroll
        for (int t = 0; t < 4; t++)
#pragma unroll
            for (int r = 0; r < 8; r++) o[t][r] *= alpha[r];

        // intra-wave LDS RAW: wait on DS counter before re-reading P as A-fragment
        asm volatile("s_wait_dscnt 0" ::: "memory");

        const u16* pp = &P[wave][nn * LDSTR];
        v16bf pf = mk_frag(*(const u32x4*)(pp + kbA), *(const u32x4*)(pp + kbA + 16));

        // ---- O += P V  (vT rows are contiguous in keys) ----
#pragma unroll
        for (int t = 0; t < 4; t++) {
            const u16* vp = vb + (size_t)(16 * t + nn) * NQ + j0;
            v16bf vf = mk_frag(*(const u32x4*)(vp + kbA), *(const u32x4*)(vp + kbA + 16));
            o[t] = __builtin_amdgcn_wmma_f32_16x16x32_bf16(false, pf, false, vf, (short)0, o[t], false, false);
        }
    }

    // normalize + write bf16 [B*N, C] for projection GEMM
    float inv[8];
#pragma unroll
    for (int r = 0; r < 8; r++) inv[r] = 1.0f / rsum[r];
#pragma unroll
    for (int t = 0; t < 4; t++)
#pragma unroll
        for (int r = 0; r < 8; r++) {
            int m = r + 8 * hh;
            int row = b * NQ + m0 + m;
            attn_out[(size_t)row * CDIM + h * HD + 16 * t + nn] = f2bf(o[t][r] * inv[r]);
        }
}

// ---------------- host side ----------------
extern "C" void kernel_launch(void* const* d_in, const int* in_sizes, int n_in,
                              void* d_out, int out_size, void* d_ws, size_t ws_size,
                              hipStream_t stream) {
    const float* x      = (const float*)d_in[0];
    const float* qkv_w  = (const float*)d_in[1];
    const float* qkv_b  = (const float*)d_in[2];
    const float* proj_w = (const float*)d_in[3];
    const float* proj_b = (const float*)d_in[4];
    float* out = (float*)d_out;

    const size_t M = (size_t)BATCH * NQ;            // 4096
    u16* ws = (u16*)d_ws;
    u16* x_bf    = ws;                               // 4096*1024
    u16* wqkv_bf = x_bf    + M * CDIM;               // 3072*1024
    u16* wproj_bf= wqkv_bf + (size_t)3 * CDIM * CDIM;// 1024*1024
    u16* q_bf    = wproj_bf+ (size_t)CDIM * CDIM;    // [B,H,N,D]
    u16* k_bf    = q_bf    + M * CDIM;
    u16* vt_bf   = k_bf    + M * CDIM;               // [B,H,D,N]
    u16* attn_bf = x_bf;                             // reuse x_bf after QKV GEMM

    // 1) convert fp32 inputs to bf16
    {
        int n8 = (int)(M * CDIM / 8);
        cvt_f32_bf16<<<(n8 + 255) / 256, 256, 0, stream>>>(x, x_bf, n8);
        n8 = 3 * CDIM * CDIM / 8;
        cvt_f32_bf16<<<(n8 + 255) / 256, 256, 0, stream>>>(qkv_w, wqkv_bf, n8);
        n8 = CDIM * CDIM / 8;
        cvt_f32_bf16<<<(n8 + 255) / 256, 256, 0, stream>>>(proj_w, wproj_bf, n8);
    }

    // 2) QKV projection: [4096,1024] x [3072,1024]^T, scatter to q/k/vT
    {
        dim3 grid(3 * CDIM / GB_N, (unsigned)(M / GB_M));
        gemm_wmma_bf16<<<grid, 256, 0, stream>>>(x_bf, wqkv_bf, qkv_b, CDIM, 1,
                                                 nullptr, q_bf, k_bf, vt_bf);
    }

    // 3) sliding-window causal ALiBi attention
    attn_wmma<<<BATCH * NH * (NQ / 64), 128, 0, stream>>>(q_bf, k_bf, vt_bf, attn_bf);

    // 4) output projection: [4096,1024] x [1024,1024]^T -> f32 out
    {
        dim3 grid(CDIM / GB_N, (unsigned)(M / GB_M));
        gemm_wmma_bf16<<<grid, 256, 0, stream>>>(attn_bf, wproj_bf, proj_b, CDIM, 2,
                                                 out, nullptr, nullptr, nullptr);
    }
    (void)in_sizes; (void)n_in; (void)out_size; (void)ws_size;
}
